// Acoustic_Encoder_Residual_Block_58952721105301
// MI455X (gfx1250) — matrix operand order repaired and checked by reading, not compiled, not run on hardware
//
#include <hip/hip_runtime.h>
#include <hip/hip_bf16.h>
#include <math.h>

// ---------------------------------------------------------------------------
// Types / constants
// ---------------------------------------------------------------------------
typedef __attribute__((ext_vector_type(16))) __bf16 v16bf;
typedef __attribute__((ext_vector_type(8)))  float  v8f;

union Frag {
    uint4 q[2];   // 32 bytes
    v16bf v;      // 16 bf16 = 32 bytes
};

constexpr int B_ = 16;
constexpr int C_ = 512;
constexpr int T_ = 4096;

static __device__ __forceinline__ unsigned short f2bf(float f) {
    unsigned u = __float_as_uint(f);
    unsigned r = u + 0x7fffu + ((u >> 16) & 1u);   // round-to-nearest-even
    return (unsigned short)(r >> 16);
}

static __device__ __forceinline__ double bessel_i0(double x) {
    double s = 1.0, term = 1.0;
    double hx = 0.5 * x;
    for (int k = 1; k <= 30; k++) {
        double r = hx / (double)k;
        term *= r * r;
        s += term;
    }
    return s;
}

// ---------------------------------------------------------------------------
// Weight-norm + bf16 pack:  Wpk[conv][tap][co][ci]
// grid = 6*512 blocks, 256 threads. conv = blockIdx.x>>9, co = blockIdx.x&511
// ---------------------------------------------------------------------------
__global__ void __launch_bounds__(256)
prep_weights_kernel(const float* __restrict__ in_v,  const float* __restrict__ in_g,
                    const float* __restrict__ out_v, const float* __restrict__ out_g,
                    unsigned short* __restrict__ Wpk) {
    int conv_id = blockIdx.x >> 9;
    int co      = blockIdx.x & 511;
    int i       = conv_id >> 1;
    int which   = conv_id & 1;
    const float* vbase = (which ? out_v : in_v) + ((size_t)(i * C_ + co)) * (C_ * 3);
    float g = (which ? out_g : in_g)[i * C_ + co];

    int tid = threadIdx.x;
    float ss = 0.f;
    for (int idx = tid; idx < C_ * 3; idx += 256) {
        float v = vbase[idx];
        ss += v * v;
    }
    __shared__ float red[256];
    red[tid] = ss;
    __syncthreads();
    for (int s = 128; s > 0; s >>= 1) {
        if (tid < s) red[tid] += red[tid + s];
        __syncthreads();
    }
    float scale = g / sqrtf(red[0]);

    unsigned short* wout = Wpk + (size_t)conv_id * 3 * C_ * C_;
    for (int idx = tid; idx < C_ * 3; idx += 256) {
        int ci = idx / 3;
        int k  = idx - 3 * ci;
        float w = vbase[idx] * scale;
        wout[((size_t)(k * C_ + co)) * C_ + ci] = f2bf(w);
    }
}

// ---------------------------------------------------------------------------
// Fused alias-free snake activation:
//   out_bf16[b][t][c] = mask * downsample2x( snake( upsample2x(x) ) )
// Exact polyphase of the reference. Tile: 64 channels x 32 time per block.
// ---------------------------------------------------------------------------
__global__ void __launch_bounds__(256)
act_kernel(const float* __restrict__ x, const float* __restrict__ mask,
           const float* __restrict__ alpha, const float* __restrict__ beta,
           unsigned short* __restrict__ out) {
    __shared__ float xs[64][45];          // x halo tile: 32 + 12 taps, padded
    __shared__ unsigned short os[32][66]; // result tile for transposed store
    __shared__ float aa[12];              // kaiser-sinc AA filter
    __shared__ float sa[64], sib[64];     // exp(alpha), 1/(exp(beta)+1e-5)

    int tid = threadIdx.x;
    int b  = blockIdx.z;
    int c0 = blockIdx.y * 64;
    int t0 = blockIdx.x * 32;

    if (tid == 0) {
        // kaiser_sinc_filter1d(cutoff=0.25, half_width=0.3, kernel=12)
        const double half_size = 6.0, delta_f = 4.0 * 0.3;
        double A = 2.285 * (half_size - 1.0) * M_PI * delta_f + 7.95;
        double bk;
        if (A > 50.0)       bk = 0.1102 * (A - 8.7);
        else if (A >= 21.0) bk = 0.5842 * pow(A - 21.0, 0.4) + 0.07886 * (A - 21.0);
        else                bk = 0.0;
        double i0b = bessel_i0(bk);
        double f[12], sum = 0.0;
        for (int n = 0; n < 12; n++) {
            double r = 2.0 * (double)n / 11.0 - 1.0;
            double w = bessel_i0(bk * sqrt(fmax(0.0, 1.0 - r * r))) / i0b;
            double tmv = (double)n - 6.0 + 0.5;
            double sx  = 0.5 * tmv;                     // 2*cutoff*time
            double snc = sin(M_PI * sx) / (M_PI * sx);  // np.sinc
            f[n] = 0.5 * w * snc;                       // 2*cutoff*window*sinc
            sum += f[n];
        }
        for (int n = 0; n < 12; n++) aa[n] = (float)(f[n] / sum);
    }
    if (tid < 64) {
        sa[tid]  = expf(alpha[c0 + tid]);
        sib[tid] = 1.f / (expf(beta[c0 + tid]) + 1e-5f);
    }
    for (int idx = tid; idx < 64 * 44; idx += 256) {
        int c = idx / 44, lt = idx % 44;
        int g = t0 - 6 + lt;
        g = min(max(g, 0), T_ - 1);                     // edge-replicate pad
        xs[c][lt] = x[((size_t)b * C_ + c0 + c) * T_ + g];
    }
    __syncthreads();

    for (int idx = tid; idx < 64 * 32; idx += 256) {
        int c = idx >> 5, t = idx & 31;
        int tg = t0 + t;
        float a = sa[c], ib = sib[c];
        float acc = 0.f;
#pragma unroll
        for (int j = 0; j < 12; j++) {
            int u = 2 * tg - 5 + j;
            u = min(max(u, 0), 2 * T_ - 1);             // edge pad of upsampled
            int s = u >> 1, p = u & 1;
            int base = s - t0 + 6;                      // local index of x[s]
            float up;
            if (p == 0) {
                up = aa[11] * xs[c][base - 3] + aa[9] * xs[c][base - 2]
                   + aa[7]  * xs[c][base - 1] + aa[5] * xs[c][base]
                   + aa[3]  * xs[c][base + 1] + aa[1] * xs[c][base + 2];
            } else {
                up = aa[10] * xs[c][base - 2] + aa[8] * xs[c][base - 1]
                   + aa[6]  * xs[c][base]     + aa[4] * xs[c][base + 1]
                   + aa[2]  * xs[c][base + 2] + aa[0] * xs[c][base + 3];
            }
            up *= 2.f;                                  // RATIO scale
            float sn = sinf(up * a);
            acc += aa[j] * (up + ib * sn * sn);         // snake_log + down tap
        }
        acc *= mask[(size_t)b * T_ + tg];
        os[t][c] = f2bf(acc);
    }
    __syncthreads();

    for (int idx = tid; idx < 32 * 64; idx += 256) {    // channels-last store
        int t = idx >> 6, c = idx & 63;
        out[((size_t)b * T_ + t0 + t) * C_ + c0 + c] = os[t][c];
    }
}

// ---------------------------------------------------------------------------
// WMMA conv1d:  Y[b,co,t] = sum_ci,k W[k,co,ci] * A[b, t + d*(k-1), ci] + bias
// Wave tile: 32 (co) x 64 (t). WG = 8 waves -> 64 x 256 tile.
// A-frag (weights) per ISA 16-bit A 16x32 layout; B-frag is two 16B loads
// (same k order) from the channels-last bf16 activation tensor.
// ---------------------------------------------------------------------------
__global__ void __launch_bounds__(256)
conv_wmma_kernel(const unsigned short* __restrict__ Aact,
                 const unsigned short* __restrict__ W,
                 const float* __restrict__ bias,
                 const float* __restrict__ residual,
                 const float* __restrict__ fmask,
                 float* __restrict__ out,
                 int dilation) {
    int tid  = threadIdx.x;
    int lane = tid & 31;
    int wid  = tid >> 5;
    int b    = blockIdx.z;
    int co_wave = blockIdx.y * 64 + (wid >> 2) * 32;
    int t_wave  = blockIdx.x * 256 + (wid & 3) * 64;
    int m = lane & 15;      // row (A) / time col (B,C)
    int h = lane >> 4;      // lane half

    v8f acc[2][4] = {};
    const unsigned short* abase = Aact + (size_t)b * T_ * C_;

    for (int ci0 = 0; ci0 < C_; ci0 += 32) {
#pragma unroll
        for (int tap = 0; tap < 3; tap++) {
            int shift = dilation * (tap - 1);
            // A fragments (weights): lane m holds row co, K runs 8h..8h+7 and 16+8h..+7
            const unsigned short* w0 =
                W + ((size_t)(tap * C_ + co_wave + m)) * C_ + ci0 + 8 * h;
            Frag a0, a1;
            a0.q[0] = *(const uint4*)(w0);
            a0.q[1] = *(const uint4*)(w0 + 16);
            a1.q[0] = *(const uint4*)(w0 + 16 * C_);
            a1.q[1] = *(const uint4*)(w0 + 16 * C_ + 16);
#pragma unroll
            for (int j = 0; j < 4; j++) {
                int t = t_wave + j * 16 + m + shift;
                Frag bf;
                if ((unsigned)t < (unsigned)T_) {
                    const uint4* p =
                        (const uint4*)(abase + (size_t)t * C_ + ci0 + 8 * h);
                    bf.q[0] = p[0];
                    bf.q[1] = p[2];
                } else {                                 // zero 'SAME' padding
                    bf.q[0] = make_uint4(0u, 0u, 0u, 0u);
                    bf.q[1] = make_uint4(0u, 0u, 0u, 0u);
                }
                acc[0][j] = __builtin_amdgcn_wmma_f32_16x16x32_bf16(
                    false, a0.v, false, bf.v, (short)0, acc[0][j], false, false);
                acc[1][j] = __builtin_amdgcn_wmma_f32_16x16x32_bf16(
                    false, a1.v, false, bf.v, (short)0, acc[1][j], false, false);
            }
        }
    }

    // Epilogue: C/D layout -> element e: M = 8*h + e, N = m
#pragma unroll
    for (int mi = 0; mi < 2; mi++) {
#pragma unroll
        for (int j = 0; j < 4; j++) {
#pragma unroll
            for (int e = 0; e < 8; e++) {
                int co = co_wave + mi * 16 + 8 * h + e;
                int t  = t_wave + j * 16 + m;
                float v = acc[mi][j][e] + bias[co];
                size_t oi = ((size_t)b * C_ + co) * T_ + t;
                if (residual) v += residual[oi];
                if (fmask)    v *= fmask[(size_t)b * T_ + t];
                out[oi] = v;
            }
        }
    }
}

// ---------------------------------------------------------------------------
// Launch
// ---------------------------------------------------------------------------
extern "C" void kernel_launch(void* const* d_in, const int* in_sizes, int n_in,
                              void* d_out, int out_size, void* d_ws, size_t ws_size,
                              hipStream_t stream) {
    (void)in_sizes; (void)n_in; (void)out_size; (void)ws_size;

    const float* x      = (const float*)d_in[0];
    const float* fmask  = (const float*)d_in[1];
    const float* in_a   = (const float*)d_in[2];
    const float* in_be  = (const float*)d_in[3];
    const float* out_a  = (const float*)d_in[4];
    const float* out_be = (const float*)d_in[5];
    const float* in_v   = (const float*)d_in[6];
    const float* in_g   = (const float*)d_in[7];
    const float* in_b   = (const float*)d_in[8];
    const float* out_v  = (const float*)d_in[9];
    const float* out_g  = (const float*)d_in[10];
    const float* out_b  = (const float*)d_in[11];

    char* ws = (char*)d_ws;
    unsigned short* Wpk  = (unsigned short*)(ws);                 //  9.4 MiB
    unsigned short* Abuf = (unsigned short*)(ws + 9437184);       // 64   MiB
    float* P0 = (float*)(ws + 76546048);                          // 128  MiB
    float* P1 = (float*)(ws + 210763776);                         // 128  MiB

    prep_weights_kernel<<<dim3(6 * 512), 256, 0, stream>>>(in_v, in_g, out_v, out_g, Wpk);

    const int dil[3] = {1, 3, 5};
    const float* cur = x;
    float* Ybufs[3] = {P0, P1, P0};
    float* Outs[3]  = {P0, P1, (float*)d_out};

    dim3 actGrid(T_ / 32, C_ / 64, B_);
    dim3 convGrid(T_ / 256, C_ / 64, B_);

    for (int i = 0; i < 3; i++) {
        float* Y = Ybufs[i];
        float* O = Outs[i];
        act_kernel<<<actGrid, 256, 0, stream>>>(cur, fmask, in_a + i * C_, in_be + i * C_, Abuf);
        conv_wmma_kernel<<<convGrid, 256, 0, stream>>>(
            Abuf, Wpk + (size_t)(2 * i) * 3 * C_ * C_, in_b + i * C_,
            nullptr, nullptr, Y, dil[i]);
        act_kernel<<<actGrid, 256, 0, stream>>>(Y, fmask, out_a + i * C_, out_be + i * C_, Abuf);
        conv_wmma_kernel<<<convGrid, 256, 0, stream>>>(
            Abuf, Wpk + (size_t)(2 * i + 1) * 3 * C_ * C_, out_b + i * C_,
            cur, (i == 2) ? fmask : nullptr, O, 1);
        cur = O;
    }
}
